// net_77360950936195
// MI455X (gfx1250) — matrix operand order repaired and checked by reading, not compiled, not run on hardware
//
#include <hip/hip_runtime.h>
#include <hip/hip_bf16.h>
#include <math.h>
#include <stdint.h>

// ---------------------------------------------------------------------------
// Types for CDNA5 WMMA (gfx1250, wave32)
// ---------------------------------------------------------------------------
typedef __attribute__((ext_vector_type(16))) __bf16 v16bf;
typedef __attribute__((ext_vector_type(8)))  float  v8f;

union FragBF {
    v16bf        v;
    unsigned int u[8];
};

// bf16 helpers (storage = raw ushort bits)
__device__ __forceinline__ unsigned short f2bf(float f) {
    union { float f; unsigned int u; } x; x.f = f;
    unsigned int r = x.u + 0x7FFFu + ((x.u >> 16) & 1u);   // round-nearest-even
    return (unsigned short)(r >> 16);
}
__device__ __forceinline__ float bf2f(unsigned short b) {
    union { unsigned int u; float f; } x; x.u = ((unsigned int)b) << 16;
    return x.f;
}

// ---------------------------------------------------------------------------
// fp32 -> bf16 elementwise conversion (input activations)
// ---------------------------------------------------------------------------
__global__ void cvt_f32_bf16_kernel(const float* __restrict__ in,
                                    unsigned short* __restrict__ out, int n) {
    int i = blockIdx.x * blockDim.x + threadIdx.x;
    if (i < n) out[i] = f2bf(in[i]);
}

// ---------------------------------------------------------------------------
// Weight repack: fp32 [Cout][K] -> bf16 [CoutPad][Kpad], zero padded
// ---------------------------------------------------------------------------
__global__ void wcvt_kernel(const float* __restrict__ w,
                            unsigned short* __restrict__ out,
                            int Cout, int K, int Kpad, int CoutPad) {
    int idx = blockIdx.x * blockDim.x + threadIdx.x;
    int total = CoutPad * Kpad;
    if (idx >= total) return;
    int m = idx / Kpad;
    int k = idx - m * Kpad;
    unsigned short v = 0;
    if (m < Cout && k < K) v = f2bf(w[(size_t)m * K + k]);
    out[idx] = v;
}

// ---------------------------------------------------------------------------
// Implicit-GEMM conv with V_WMMA_F32_16X16X32_BF16
//   M = Cout, N = B*Hout*Wout, K = Cin*KSZ*KSZ (Kpad = ceil32)
//   block = 256 threads = 8 waves; C-tile 64(M) x 128(N); each wave: 32Mx32N
//   A (weight) tile loaded with global_load_async_to_lds_b128 (ASYNCcnt path).
// ---------------------------------------------------------------------------
#define LDS_STRIDE 48   // 32 K-elements + pad (96B rows: 16B aligned, bank-staggered)

template <int KSZ>
__global__ __launch_bounds__(256)
void conv_wmma_kernel(const unsigned short* __restrict__ actIn,   // bf16 [B][Cin][Hin][Win]
                      const unsigned short* __restrict__ wbf,     // bf16 [CoutPad][Kpad]
                      const float* __restrict__ bias,             // fp32 [Cout]
                      unsigned short* __restrict__ actOut,        // bf16 [B][Cout][Hout][Wout]
                      float* __restrict__ outF32,                 // fp32 (last layer)
                      int Bn, int Cin, int Hin, int Win,
                      int Cout, int Hout, int Wout,
                      int pad, int K, int Kpad, int storeF32) {
    __shared__ unsigned short Alds[64 * LDS_STRIDE];    //  6 KB
    __shared__ unsigned short Blds[128 * LDS_STRIDE];   // 12 KB

    const int Ntot = Bn * Hout * Wout;
    const int n_tile = blockIdx.x * 128;
    const int m_tile = blockIdx.y * 64;

    const int tid    = threadIdx.x;
    const int wave   = tid >> 5;
    const int lane   = tid & 31;
    const int lane16 = lane & 15;
    const int half   = lane >> 4;
    const int m_off  = (wave & 1) * 32;    // wave's M offset within tile
    const int n_off  = (wave >> 1) * 32;   // wave's N offset within tile

    // --- A-tile cooperative-fill coords: 64 rows x 32 cols, 8 bf16/lane ----
    const int arow = tid >> 2;             // 0..63
    const int acol = (tid & 3) * 8;        // 0,8,16,24
    const unsigned int ldsA =
        (unsigned int)(uintptr_t)&Alds[arow * LDS_STRIDE + acol];   // 16B aligned

    // --- B-tile cooperative-fill coords: 128 n-rows x 32 k, 16 bf16/thread -
    const int nloc  = tid >> 1;            // 0..127
    const int kbase = (tid & 1) * 16;      // 0 or 16
    const int ngl   = n_tile + nloc;
    const bool valid_n = (ngl < Ntot);
    int oxn = 0, oyn = 0, bn = 0;
    if (valid_n) {
        oxn = ngl % Wout;
        int rr = ngl / Wout;
        oyn = rr % Hout;
        bn  = rr / Hout;
    }

    v8f acc00, acc01, acc10, acc11;
#pragma unroll
    for (int i = 0; i < 8; ++i) {
        acc00[i] = 0.0f; acc01[i] = 0.0f; acc10[i] = 0.0f; acc11[i] = 0.0f;
    }

    for (int kk = 0; kk < Kpad; kk += 32) {
        // ---- A tile: async global->LDS copy, 16B per lane (4KB total) ----
        {
            const unsigned long long ga = (unsigned long long)(uintptr_t)
                (wbf + (size_t)(m_tile + arow) * Kpad + kk + acol);
            asm volatile("global_load_async_to_lds_b128 %0, %1, off"
                         :: "v"(ldsA), "v"(ga) : "memory");
        }
        // ---- B tile: im2col gather, stored [n][k] ------------------------
#pragma unroll 4
        for (int i = 0; i < 16; ++i) {
            const int klocal = kbase + i;
            const int kg = kk + klocal;
            unsigned short v = 0;
            if (valid_n && kg < K) {
                int ci, ky, kx;
                if (KSZ == 1) { ci = kg; ky = 0; kx = 0; }
                else {
                    ci = kg / (KSZ * KSZ);
                    int r = kg - ci * (KSZ * KSZ);
                    ky = r / KSZ;
                    kx = r - ky * KSZ;
                }
                const int iy = oyn + ky - pad;
                const int ix = oxn + kx - pad;
                if (iy >= 0 && iy < Hin && ix >= 0 && ix < Win)
                    v = actIn[(((size_t)bn * Cin + ci) * Hin + iy) * Win + ix];
            }
            Blds[nloc * LDS_STRIDE + klocal] = v;
        }
        asm volatile("s_wait_asynccnt 0x0" ::: "memory");
        __syncthreads();

        // ---- fragments per ISA 16-bit A/B layouts ------------------------
        FragBF a0, a1, b0, b1;
#pragma unroll
        for (int v = 0; v < 8; ++v) {
            const int kp = ((v < 4) ? (2 * v) : (16 + 2 * (v - 4))) + 8 * half;
            const int kb = ((v < 4) ? (2 * v) : (16 + 2 * (v - 4))) + 8 * half;
            a0.u[v] = *(const unsigned int*)&Alds[(m_off +      lane16) * LDS_STRIDE + kp];
            a1.u[v] = *(const unsigned int*)&Alds[(m_off + 16 + lane16) * LDS_STRIDE + kp];
            b0.u[v] = *(const unsigned int*)&Blds[(n_off +      lane16) * LDS_STRIDE + kb];
            b1.u[v] = *(const unsigned int*)&Blds[(n_off + 16 + lane16) * LDS_STRIDE + kb];
        }
        acc00 = __builtin_amdgcn_wmma_f32_16x16x32_bf16(false, a0.v, false, b0.v,
                                                        (short)0, acc00, false, false);
        acc01 = __builtin_amdgcn_wmma_f32_16x16x32_bf16(false, a0.v, false, b1.v,
                                                        (short)0, acc01, false, false);
        acc10 = __builtin_amdgcn_wmma_f32_16x16x32_bf16(false, a1.v, false, b0.v,
                                                        (short)0, acc10, false, false);
        acc11 = __builtin_amdgcn_wmma_f32_16x16x32_bf16(false, a1.v, false, b1.v,
                                                        (short)0, acc11, false, false);
        __syncthreads();
    }

    // ---- epilogue: bias add + store (bf16 NCHW, or fp32 for last layer) --
    const size_t HW = (size_t)Hout * Wout;
#pragma unroll
    for (int col = 0; col < 2; ++col) {
        const int n = n_tile + n_off + col * 16 + lane16;
        if (n < Ntot) {
            const int ox = n % Wout;
            int rr = n / Wout;
            const int oy = rr % Hout;
            const int b  = rr / Hout;
            const size_t base = (size_t)b * Cout * HW + (size_t)oy * Wout + ox;
            const v8f& accA = col ? acc01 : acc00;
            const v8f& accB = col ? acc11 : acc10;
#pragma unroll
            for (int r = 0; r < 8; ++r) {
                const int m0 = m_tile + m_off + half * 8 + r;
                const int m1 = m0 + 16;
                if (m0 < Cout) {
                    const float v = accA[r] + bias[m0];
                    if (storeF32) outF32[base + (size_t)m0 * HW] = v;
                    else          actOut[base + (size_t)m0 * HW] = f2bf(v);
                }
                if (m1 < Cout) {
                    const float v = accB[r] + bias[m1];
                    if (storeF32) outF32[base + (size_t)m1 * HW] = v;
                    else          actOut[base + (size_t)m1 * HW] = f2bf(v);
                }
            }
        }
    }
}

// ---------------------------------------------------------------------------
// 2x2 stride-2 maxpool with -inf padding, bf16 NCHW
// ---------------------------------------------------------------------------
__global__ void pool_kernel(const unsigned short* __restrict__ in,
                            unsigned short* __restrict__ out,
                            int Bn, int C, int Hin, int Win,
                            int Hout, int Wout, int pad) {
    const int idx = blockIdx.x * blockDim.x + threadIdx.x;
    const int total = Bn * C * Hout * Wout;
    if (idx >= total) return;
    const int x = idx % Wout;
    int r = idx / Wout;
    const int y = r % Hout; r /= Hout;
    const int c = r % C;
    const int b = r / C;
    float m = -INFINITY;
#pragma unroll
    for (int dy = 0; dy < 2; ++dy)
#pragma unroll
        for (int dx = 0; dx < 2; ++dx) {
            const int iy = 2 * y - pad + dy;
            const int ix = 2 * x - pad + dx;
            if (iy >= 0 && iy < Hin && ix >= 0 && ix < Win)
                m = fmaxf(m, bf2f(in[(((size_t)b * C + c) * Hin + iy) * Win + ix]));
        }
    out[idx] = f2bf(m);
}

// ---------------------------------------------------------------------------
// YOLO head. feat: fp32 [8][6][13][13].  out: [8*13*13*6] x_out ++ [1] loss.
// The reference's serial scan parallelizes: the per-cell cls state after step t
// is f^{k_t}(s0) with f(v)=v+softmax(v) and k_t = #high cells with index < t.
// ---------------------------------------------------------------------------
#define GDIM 13
__global__ __launch_bounds__(256)
void head_kernel(const float* __restrict__ feat, float* __restrict__ out) {
    const int Bn = 8, T = Bn * GDIM * GDIM;               // 1352
    __shared__ unsigned char highs[Bn * GDIM * GDIM];
    __shared__ int pref[Bn * GDIM * GDIM];
    __shared__ int ktot_s;
    __shared__ float lsum[256];

    const float gxmin = 0.567f - 0.206f * 0.5f, gxmax = 0.567f + 0.206f * 0.5f;
    const float gymin = 0.469f - 0.53f  * 0.5f, gymax = 0.469f + 0.53f  * 0.5f;
    const int tid = threadIdx.x;

    // pass 1: high flags (depend only on channels 2..5)
    for (int t = tid; t < T; t += 256) {
        const int k = t % GDIM;
        int r = t / GDIM;
        const int j = r % GDIM;
        const int b = r / GDIM;
        const float* fb = feat + (size_t)b * 6 * GDIM * GDIM;
        const float tx = fb[2 * GDIM * GDIM + k * GDIM + j];
        const float ty = fb[3 * GDIM * GDIM + k * GDIM + j];
        const float tw = fb[4 * GDIM * GDIM + k * GDIM + j];
        const float th = fb[5 * GDIM * GDIM + k * GDIM + j];
        const float px = ((float)j + tx) / (float)GDIM;
        const float py = ((float)k + ty) / (float)GDIM;
        const float pw = __expf(tw) * 0.2f / (float)GDIM;
        const float ph = __expf(th) * 0.5f / (float)GDIM;
        const float pxmin = px - pw * 0.5f, pxmax = px + pw * 0.5f;
        const float pymin = py - ph * 0.5f, pymax = py + ph * 0.5f;
        const float xmin = fmaxf(pxmin, gxmin), xmax = fminf(pxmax, gxmax);
        const float ymin = fmaxf(pymin, gymin), ymax = fminf(pymax, gymax);
        const float iou_area = (xmax - xmin) * (ymax - ymin);
        const float iou_pred = (pxmax - pxmin) * (pymax - pymin);
        const float iou_gt   = (gxmax - gxmin) * (pymax - pymin);   // reference bug kept
        const bool zero = (pxmin > gxmax) || (pxmax < gxmin) ||
                          (pymin > gymax) || (pxmax < gymin);       // reference bug kept
        const float iou = zero ? 0.0f : iou_area / (iou_gt + iou_pred - iou_area);
        highs[t] = (iou > 0.7f) ? 1 : 0;
    }
    __syncthreads();

    // pass 2: prefix count of highs (tiny, serial on one lane)
    if (tid == 0) {
        int run = 0;
        for (int t = 0; t < T; ++t) { pref[t] = run; run += highs[t]; }
        ktot_s = run;
    }
    __syncthreads();
    const int Ktot = ktot_s;

    // pass 3: iterated in-place softmax updates + loss contributions
    float lacc = 0.0f;
    for (int t = tid; t < T; t += 256) {
        const int k = t % GDIM;
        int r = t / GDIM;
        const int j = r % GDIM;
        const int b = r / GDIM;
        const float* fb = feat + (size_t)b * 6 * GDIM * GDIM;
        const float tx = fb[2 * GDIM * GDIM + k * GDIM + j];
        const float ty = fb[3 * GDIM * GDIM + k * GDIM + j];
        const float tw = fb[4 * GDIM * GDIM + k * GDIM + j];
        const float th = fb[5 * GDIM * GDIM + k * GDIM + j];
        const float px = ((float)j + tx) / (float)GDIM;
        const float py = ((float)k + ty) / (float)GDIM;
        const float pw = __expf(tw) * 0.2f / (float)GDIM;
        const float ph = __expf(th) * 0.5f / (float)GDIM;
        const float pxmin = px - pw * 0.5f, pxmax = px + pw * 0.5f;
        const float pymin = py - ph * 0.5f, pymax = py + ph * 0.5f;
        const float xmin = fmaxf(pxmin, gxmin), xmax = fminf(pxmax, gxmax);
        const float ymin = fmaxf(pymin, gymin), ymax = fminf(pymax, gymax);
        const float iou_area = (xmax - xmin) * (ymax - ymin);
        const float iou_pred = (pxmax - pxmin) * (pymax - pymin);
        const float iou_gt   = (gxmax - gxmin) * (pymax - pymin);
        const bool zero = (pxmin > gxmax) || (pxmax < gxmin) ||
                          (pymin > gymax) || (pxmax < gymin);
        const float iou = zero ? 0.0f : iou_area / (iou_gt + iou_pred - iou_area);
        const bool low  = iou < 0.3f;
        const bool high = iou > 0.7f;
        const float absl = fabsf(0.567f - tx) + fabsf(0.469f - ty) +
                           fabsf(0.206f - tw) + fabsf(0.53f - th);

        float v0 = fb[0 * GDIM * GDIM + k * GDIM + j];
        float v1 = fb[1 * GDIM * GDIM + k * GDIM + j];
        const int kt = pref[t];
        for (int i = 0; i < kt; ++i) {
            const float s1 = 1.0f / (1.0f + __expf(v0 - v1));
            v0 += 1.0f - s1; v1 += s1;
        }
        const float sm1 = 1.0f / (1.0f + __expf(v0 - v1));
        const float sm0 = 1.0f - sm1;
        if (low)  lacc += 1.0f - sm0;
        if (high) lacc += (1.0f - (v1 + sm1)) + absl;
        for (int i = kt; i < Ktot; ++i) {
            const float s1 = 1.0f / (1.0f + __expf(v0 - v1));
            v0 += 1.0f - s1; v1 += s1;
        }
        float* o = out + (size_t)t * 6;
        o[0] = v0; o[1] = v1; o[2] = tx; o[3] = ty; o[4] = tw; o[5] = th;
    }
    lsum[tid] = lacc;
    __syncthreads();
    for (int s = 128; s > 0; s >>= 1) {
        if (tid < s) lsum[tid] += lsum[tid + s];
        __syncthreads();
    }
    if (tid == 0) out[T * 6] = lsum[0];
}

// ---------------------------------------------------------------------------
// Host launcher
// ---------------------------------------------------------------------------
extern "C" void kernel_launch(void* const* d_in, const int* in_sizes, int n_in,
                              void* d_out, int out_size, void* d_ws, size_t ws_size,
                              hipStream_t stream) {
    (void)in_sizes; (void)n_in; (void)out_size; (void)ws_size;

    static const int CO[18] = {32,64,128,64,128,256,128,256,512,256,512,256,512,1024,512,1024,512,6};
    static const int CI[18] = {3,32,64,128,64,128,256,128,256,512,256,512,256,512,1024,512,1024,512};
    static const int KS[18] = {3,3,3,1,3,3,1,3,3,1,3,1,3,3,1,3,1,3};
    static const int PD[18] = {0,0,1,0,1,1,0,0,1,0,1,0,0,1,0,1,0,1};
    // ops: >=0 conv idx; -1 pool pad=1; -2 pool pad=0
    static const int OPS_[23] = {0,-1,1,-1,2,3,4,-2,5,6,7,-1,8,9,10,11,12,-1,13,14,15,16,17};
    const int BATCH = 8;

    // --- replay dims to size workspace ------------------------------------
    size_t maxAct = (size_t)BATCH * 3 * 416 * 416;
    {
        int H = 416, C = 3;
        for (int i = 0; i < 23; ++i) {
            if (OPS_[i] >= 0) {
                const int l = OPS_[i];
                H = H + 2 * PD[l] - KS[l] + 1;
                C = CO[l];
            } else {
                const int p = (OPS_[i] == -1) ? 1 : 0;
                H = (H + 2 * p - 2) / 2 + 1;
            }
            const size_t e = (size_t)BATCH * C * H * H;
            if (e > maxAct) maxAct = e;
        }
    }
    size_t woff[18], wtot = 0;
    int Kpads[18], CoutPads[18];
    for (int l = 0; l < 18; ++l) {
        const int K = CI[l] * KS[l] * KS[l];
        Kpads[l]    = ((K + 31) / 32) * 32;
        CoutPads[l] = ((CO[l] + 63) / 64) * 64;
        woff[l] = wtot;
        wtot += (size_t)CoutPads[l] * Kpads[l];
    }

    char* ws = (char*)d_ws;
    size_t off = 0;
    auto take = [&](size_t bytes) -> char* {
        char* p = ws + off;
        off += (bytes + 255) & ~(size_t)255;
        return p;
    };
    unsigned short* actA = (unsigned short*)take(maxAct * 2);
    unsigned short* actB = (unsigned short*)take(maxAct * 2);
    unsigned short* wbuf = (unsigned short*)take(wtot * 2);
    float*          feat = (float*)take((size_t)BATCH * 6 * 13 * 13 * 4);

    // --- 1) input fp32 -> bf16 --------------------------------------------
    {
        const int n = BATCH * 3 * 416 * 416;
        cvt_f32_bf16_kernel<<<(n + 255) / 256, 256, 0, stream>>>(
            (const float*)d_in[0], actA, n);
    }
    // --- 2) weights fp32 -> padded bf16 -----------------------------------
    for (int l = 0; l < 18; ++l) {
        const int K = CI[l] * KS[l] * KS[l];
        const int total = CoutPads[l] * Kpads[l];
        wcvt_kernel<<<(total + 255) / 256, 256, 0, stream>>>(
            (const float*)d_in[1 + 2 * l], wbuf + woff[l],
            CO[l], K, Kpads[l], CoutPads[l]);
    }
    // --- 3) backbone ------------------------------------------------------
    unsigned short* cur = actA;
    unsigned short* oth = actB;
    int H = 416, C = 3;
    for (int i = 0; i < 23; ++i) {
        if (OPS_[i] >= 0) {
            const int l = OPS_[i];
            const int Hout = H + 2 * PD[l] - KS[l] + 1;
            const int K = CI[l] * KS[l] * KS[l];
            const int Ntot = BATCH * Hout * Hout;
            const int lastLayer = (l == 17) ? 1 : 0;
            dim3 grid((Ntot + 127) / 128, CoutPads[l] / 64);
            if (KS[l] == 1) {
                conv_wmma_kernel<1><<<grid, 256, 0, stream>>>(
                    cur, wbuf + woff[l], (const float*)d_in[2 + 2 * l],
                    oth, feat,
                    BATCH, CI[l], H, H, CO[l], Hout, Hout,
                    PD[l], K, Kpads[l], lastLayer);
            } else {
                conv_wmma_kernel<3><<<grid, 256, 0, stream>>>(
                    cur, wbuf + woff[l], (const float*)d_in[2 + 2 * l],
                    oth, feat,
                    BATCH, CI[l], H, H, CO[l], Hout, Hout,
                    PD[l], K, Kpads[l], lastLayer);
            }
            H = Hout; C = CO[l];
            unsigned short* t = cur; cur = oth; oth = t;
        } else {
            const int p = (OPS_[i] == -1) ? 1 : 0;
            const int Hout = (H + 2 * p - 2) / 2 + 1;
            const int total = BATCH * C * Hout * Hout;
            pool_kernel<<<(total + 255) / 256, 256, 0, stream>>>(
                cur, oth, BATCH, C, H, H, Hout, Hout, p);
            H = Hout;
            unsigned short* t = cur; cur = oth; oth = t;
        }
    }
    // --- 4) YOLO head -----------------------------------------------------
    head_kernel<<<1, 256, 0, stream>>>(feat, (float*)d_out);
}
